// GCNIIConv_6150393168691
// MI455X (gfx1250) — matrix operand order repaired and checked by reading, not compiled, not run on hardware
//
#include <hip/hip_runtime.h>
#include <stdint.h>

typedef __attribute__((ext_vector_type(16))) __bf16 v16bf;
typedef __attribute__((ext_vector_type(8)))  float  v8f;

#define NNODE 40000
#define CH    128
#define KNEI  16
#define DEG   17
#define ALPHA 0.1f
#define BETA  0.11778303565638346f          /* ln(1.125) */
#define S1    ((1.0f - ALPHA) * (1.0f - BETA))
#define S2    (ALPHA * (1.0f - BETA))

__device__ __forceinline__ uint16_t f2bf(float f) {
  uint32_t u = __float_as_uint(f);
  u += 0x7fffu + ((u >> 16) & 1u);          // round-to-nearest-even
  return (uint16_t)(u >> 16);
}
__device__ __forceinline__ float bflo(uint32_t u) { return __uint_as_float(u << 16); }
__device__ __forceinline__ float bfhi(uint32_t u) { return __uint_as_float(u & 0xffff0000u); }

// ---- prep: M1[o][c] = beta*W1[o][c] + s1*delta ; M2 likewise with s2 (bf16, row-major o) ----
__global__ void prep_weights(const float* __restrict__ W1, const float* __restrict__ W2,
                             uint16_t* __restrict__ m1, uint16_t* __restrict__ m2) {
  int t = blockIdx.x * 256 + threadIdx.x;   // 0..16383
  int o = t >> 7, c = t & 127;
  float d = (o == c) ? 1.0f : 0.0f;
  m1[t] = f2bf(BETA * W1[t] + S1 * d);
  m2[t] = f2bf(BETA * W2[t] + S2 * d);
}

// ---- prep: transpose x [C][N] f32 -> xT [N][C] bf16 (LDS-tiled, coalesced both sides) ----
__global__ void prep_xt(const float* __restrict__ x, uint16_t* __restrict__ xT) {
  __shared__ float tile[32][33];
  int n0 = blockIdx.x * 32, c0 = blockIdx.y * 32;
  int tx = threadIdx.x, ty = threadIdx.y;
#pragma unroll
  for (int i = 0; i < 4; ++i)
    tile[ty + i * 8][tx] = x[(size_t)(c0 + ty + i * 8) * NNODE + n0 + tx];
  __syncthreads();
#pragma unroll
  for (int i = 0; i < 4; ++i)
    xT[(size_t)(n0 + ty + i * 8) * CH + c0 + tx] = f2bf(tile[tx][ty + i * 8]);
}

// ---- prep: x_0 f32 -> bf16 ----
__global__ void prep_x0(const float* __restrict__ x0, uint16_t* __restrict__ x0b) {
  int t = blockIdx.x * 256 + threadIdx.x;   // 4 elements/thread
  float4 v = ((const float4*)x0)[t];
  uint2 p;
  p.x = (uint32_t)f2bf(v.x) | ((uint32_t)f2bf(v.y) << 16);
  p.y = (uint32_t)f2bf(v.z) | ((uint32_t)f2bf(v.w) << 16);
  ((uint2*)x0b)[t] = p;
}

// ---- main: gather-mean -> dual bf16 WMMA GEMM -> bias+relu -> transposed store ----
union Frag { v16bf v; uint4 q[2]; };

__global__ __launch_bounds__(256) void gcnii_main(
    const uint16_t* __restrict__ xT,   // [N][C] bf16
    const uint16_t* __restrict__ x0b,  // [N][C] bf16
    const uint16_t* __restrict__ m1g,  // [128][128] bf16
    const uint16_t* __restrict__ m2g,
    const int*      __restrict__ eidx, // [N][16]
    const float*    __restrict__ bias, // [128]
    float*          __restrict__ out)  // [C][N]
{
  extern __shared__ unsigned char smem[];
  uint16_t* m1l = (uint16_t*)smem;          // 128 rows x 136 bf16 (padded)
  uint16_t* m2l = m1l + 128 * 136;
  uint16_t* agl = m2l + 128 * 136;          // 8 waves x 16 rows x 136 bf16

  const int tid  = threadIdx.x;
  const int wave = tid >> 5;
  const int lane = tid & 31;
  const int half = lane >> 4;
  const int l16  = lane & 15;

  // cooperative weight load into padded LDS (16B chunks)
#pragma unroll
  for (int i = 0; i < 8; ++i) {
    int k = tid + i * 256;                  // chunk of 8 bf16
    int o = k >> 4, cc = (k & 15) * 8;
    *(uint4*)(m1l + o * 136 + cc) = *(const uint4*)(m1g + o * 128 + cc);
    *(uint4*)(m2l + o * 136 + cc) = *(const uint4*)(m2g + o * 128 + cc);
  }

  const int  nb     = (blockIdx.x * 8 + wave) * 16;   // 16-node tile per wave
  const bool active = nb < NNODE;

  if (active) {
    // gather-mean: lane covers 8 channels of node (nb + 2*i2 + half)
    const int cg = l16;
    uint16_t* abase = agl + wave * 16 * 136;
    for (int i2 = 0; i2 < 8; ++i2) {
      int node = nb + i2 * 2 + half;
      float a[8] = {0.f, 0.f, 0.f, 0.f, 0.f, 0.f, 0.f, 0.f};
#pragma unroll
      for (int j = 0; j < DEG; ++j) {
        int nj = (j < KNEI) ? eidx[node * KNEI + j] : node;  // self-loop last
        uint4 u = *(const uint4*)(xT + nj * CH + cg * 8);
        a[0] += bflo(u.x); a[1] += bfhi(u.x);
        a[2] += bflo(u.y); a[3] += bfhi(u.y);
        a[4] += bflo(u.z); a[5] += bfhi(u.z);
        a[6] += bflo(u.w); a[7] += bfhi(u.w);
      }
      const float s = 1.0f / (float)DEG;
      uint4 p;
      p.x = (uint32_t)f2bf(a[0] * s) | ((uint32_t)f2bf(a[1] * s) << 16);
      p.y = (uint32_t)f2bf(a[2] * s) | ((uint32_t)f2bf(a[3] * s) << 16);
      p.z = (uint32_t)f2bf(a[4] * s) | ((uint32_t)f2bf(a[5] * s) << 16);
      p.w = (uint32_t)f2bf(a[6] * s) | ((uint32_t)f2bf(a[7] * s) << 16);
      *(uint4*)(abase + (i2 * 2 + half) * 136 + cg * 8) = p;
    }
  }
  __syncthreads();
  if (!active) return;

  // A fragments: ISA 16-bit A 16x32 layout (lane%16 = M; K runs at half*8 / 16+half*8)
  Frag aA[4], aX[4];
  const uint16_t* arow = agl + wave * 16 * 136 + l16 * 136;
  const uint16_t* xrow = x0b + (size_t)(nb + l16) * CH;
#pragma unroll
  for (int ks = 0; ks < 4; ++ks) {
    aA[ks].q[0] = *(const uint4*)(arow + ks * 32 + half * 8);
    aA[ks].q[1] = *(const uint4*)(arow + ks * 32 + 16 + half * 8);
    aX[ks].q[0] = *(const uint4*)(xrow + ks * 32 + half * 8);
    aX[ks].q[1] = *(const uint4*)(xrow + ks * 32 + 16 + half * 8);
  }

  // 8 output-channel tiles x 4 k-steps x 2 GEMMs = 64 WMMA
#pragma unroll
  for (int t = 0; t < 8; ++t) {
    v8f acc = {};
    // B 32x16 layout: lane%16 = N (out channel), K runs at half*8 / 16+half*8
    const uint16_t* w1r = m1l + (t * 16 + l16) * 136 + half * 8;
    const uint16_t* w2r = m2l + (t * 16 + l16) * 136 + half * 8;
#pragma unroll
    for (int ks = 0; ks < 4; ++ks) {
      Frag b1, b2;
      b1.q[0] = *(const uint4*)(w1r + ks * 32);
      b1.q[1] = *(const uint4*)(w1r + ks * 32 + 16);
      b2.q[0] = *(const uint4*)(w2r + ks * 32);
      b2.q[1] = *(const uint4*)(w2r + ks * 32 + 16);
      acc = __builtin_amdgcn_wmma_f32_16x16x32_bf16(false, aA[ks].v, false, b1.v,
                                                    (short)0, acc, false, false);
      acc = __builtin_amdgcn_wmma_f32_16x16x32_bf16(false, aX[ks].v, false, b2.v,
                                                    (short)0, acc, false, false);
    }
    const float bv = bias[t * 16 + l16];
    // D layout: VGPR r, lane -> node = nb + r + 8*half, outch = t*16 + lane%16
#pragma unroll
    for (int r = 0; r < 8; ++r) {
      float o = acc[r] + bv;
      o = o > 0.0f ? o : 0.0f;
      out[(size_t)(t * 16 + l16) * NNODE + nb + r + half * 8] = o;
    }
  }
}

extern "C" void kernel_launch(void* const* d_in, const int* in_sizes, int n_in,
                              void* d_out, int out_size, void* d_ws, size_t ws_size,
                              hipStream_t stream) {
  (void)in_sizes; (void)n_in; (void)out_size; (void)ws_size;
  const float* x    = (const float*)d_in[0];   // [1,128,N,1]
  const float* x0   = (const float*)d_in[1];   // [1,N,128]
  const int*   ei   = (const int*)d_in[2];     // [2,1,N,16] (only half 0 used)
  const float* W1   = (const float*)d_in[3];
  const float* W2   = (const float*)d_in[4];
  const float* bias = (const float*)d_in[5];
  float* out = (float*)d_out;

  uint16_t* xT  = (uint16_t*)d_ws;                    // N*128 bf16
  uint16_t* x0b = xT  + (size_t)NNODE * CH;           // N*128 bf16
  uint16_t* m1  = x0b + (size_t)NNODE * CH;           // 128*128 bf16
  uint16_t* m2  = m1  + CH * CH;

  prep_weights<<<64, 256, 0, stream>>>(W1, W2, m1, m2);
  prep_xt<<<dim3(NNODE / 32, CH / 32), dim3(32, 8), 0, stream>>>(x, xT);
  prep_x0<<<(NNODE * CH) / (256 * 4), 256, 0, stream>>>(x0, x0b);

  size_t shmem = (size_t)(2 * 128 * 136 + 8 * 16 * 136) * sizeof(uint16_t); // 104448 B
  gcnii_main<<<313, 256, shmem, stream>>>(xT, x0b, m1, m2, ei, bias, out);
}
